// NegativeSquare_46608985096388
// MI455X (gfx1250) — matrix operand order repaired and checked by reading, not compiled, not run on hardware
//
#include <hip/hip_runtime.h>
#include <hip/hip_bf16.h>
#include <stdint.h>

// B=8, T=1024, D=512, K=1024
// M = B*T = 8192 rows of x ; N = K = 1024 codebook rows ; reduce over D = 512.
#define M_DIM  8192
#define N_DIM  1024
#define D_DIM  512

typedef __attribute__((ext_vector_type(16))) __bf16 v16bf;
typedef __attribute__((ext_vector_type(8)))  float  v8f;
typedef __attribute__((ext_vector_type(4)))  int    v4i;

// ---------------------------------------------------------------------------
// RNE float -> bf16 pack (2 floats -> one u32, low half = first element)
// ---------------------------------------------------------------------------
__device__ __forceinline__ uint32_t pack_bf16(float a, float b) {
    uint32_t ua = __float_as_uint(a);
    ua += 0x7FFFu + ((ua >> 16) & 1u);
    uint32_t ub = __float_as_uint(b);
    ub += 0x7FFFu + ((ub >> 16) & 1u);
    return (ua >> 16) | (ub & 0xFFFF0000u);
}

// ---------------------------------------------------------------------------
// Async global->LDS copy support (CDNA5). Fallback: load + ds_store.
// Probe round 2 showed the builtin expects generic `int4*` pointers.
// ---------------------------------------------------------------------------
#if __has_builtin(__builtin_amdgcn_global_load_async_to_lds_b128)
#define HAVE_ASYNC_LDS 1
#else
#define HAVE_ASYNC_LDS 0
#endif

__device__ __forceinline__ void wait_async_copies() {
#if HAVE_ASYNC_LDS
#if __has_builtin(__builtin_amdgcn_s_wait_asynccnt)
    __builtin_amdgcn_s_wait_asynccnt(0);
#else
    asm volatile("s_wait_asynccnt 0" ::: "memory");
#endif
#endif
}

// ---------------------------------------------------------------------------
// Kernel 1 (pre-pass): per row, compute squared norm AND write bf16 copy.
// rows [0,8192) -> x ; rows [8192,9216) -> codebook. One wave32 per row.
// ---------------------------------------------------------------------------
__global__ __launch_bounds__(256) void prep_kernel(const float* __restrict__ x,
                                                   const float* __restrict__ cb,
                                                   uint32_t* __restrict__ xbf,
                                                   uint32_t* __restrict__ cbbf,
                                                   float* __restrict__ xsq,
                                                   float* __restrict__ csq) {
    const int wave = threadIdx.x >> 5;
    const int lane = threadIdx.x & 31;
    const int row  = blockIdx.x * 8 + wave;

    const float* src;
    uint32_t* dstb;   // 256 u32 per row (512 bf16)
    float* dstn;
    if (row < M_DIM) {
        src  = x + (size_t)row * D_DIM;
        dstb = xbf + (size_t)row * (D_DIM / 2);
        dstn = xsq + row;
    } else {
        const int r = row - M_DIM;
        src  = cb + (size_t)r * D_DIM;
        dstb = cbbf + (size_t)r * (D_DIM / 2);
        dstn = csq + r;
    }
    const float4* s4 = (const float4*)src;
    float sum = 0.0f;
#pragma unroll
    for (int j = 0; j < 4; ++j) {
        const int slot = j * 32 + lane;       // 0..127 float4s of the row
        float4 v = s4[slot];
        sum += v.x * v.x + v.y * v.y + v.z * v.z + v.w * v.w;
        uint2 p;
        p.x = pack_bf16(v.x, v.y);
        p.y = pack_bf16(v.z, v.w);
        *(uint2*)&dstb[slot * 2] = p;
    }
#pragma unroll
    for (int off = 16; off >= 1; off >>= 1)
        sum += __shfl_xor(sum, off, 32);
    if (lane == 0) *dstn = sum;
}

// ---------------------------------------------------------------------------
// Kernel 2: bf16 WMMA GEMM with fused distance epilogue.
//   out[m,n] = prec * (2 * sum_d x[m,d]*cb[n,d] - xsq[m] - csq[n])
// Block = 256 threads (8 waves), tile = 128(M) x 128(N), K-step = 32.
// Wave grid 2(M) x 4(N): each wave -> 64x32 = 4x2 WMMA 16x16 tiles.
// LDS: double-buffered bf16 tiles, row pitch 80B (20 u32) for bank spread.
// Tiles staged with global_load_async_to_lds_b128 (ASYNCcnt) when available.
// ---------------------------------------------------------------------------
#define PITCH_U32 20                    // 32 bf16 = 64B data + 16B pad
#define TILE_U32  (128 * PITCH_U32)     // 2560 u32 per tile buffer

__global__ __launch_bounds__(256) void vq_score_kernel(const uint32_t* __restrict__ xbf,
                                                       const uint32_t* __restrict__ cbbf,
                                                       const float* __restrict__ xsq,
                                                       const float* __restrict__ csq,
                                                       const float* __restrict__ prec_p,
                                                       float* __restrict__ out) {
    __shared__ __align__(16) uint32_t ldsU[4 * TILE_U32]; // A[2], B[2] = 40 KB

    const int tid   = threadIdx.x;
    const int lane  = tid & 31;
    const int wave  = tid >> 5;
    const int l16   = lane & 15;
    const int hiH   = (lane >> 4) & 1;      // 0: lanes 0-15, 1: lanes 16-31
    const int waveM = wave & 1;             // 2 waves along M (64 rows each)
    const int waveN = wave >> 1;            // 4 waves along N (32 cols each)

    const int m0 = blockIdx.y * 128;        // global row base of this tile
    const int n0 = blockIdx.x * 128;        // global col base of this tile

    // bf16 rows: 512 bf16 = 64 uint4 per row
    const uint4* __restrict__ xg = (const uint4*)xbf;
    const uint4* __restrict__ cg = (const uint4*)cbbf;

    v8f acc[4][2];
#pragma unroll
    for (int s = 0; s < 4; ++s)
#pragma unroll
        for (int t = 0; t < 2; ++t)
            acc[s][t] = (v8f){0.f, 0.f, 0.f, 0.f, 0.f, 0.f, 0.f, 0.f};

    // ---- stage a K-slice (kk) of both tiles into LDS buffer `buf` ----------
    // Tile = 128 rows x 32 bf16 = 512 uint4 ; 256 threads -> 2 uint4 each.
    auto copy_tile = [&](int buf, int kk) {
        uint32_t* A = &ldsU[buf * TILE_U32];
        uint32_t* B = &ldsU[2 * TILE_U32 + buf * TILE_U32];
#pragma unroll
        for (int i = 0; i < 2; ++i) {
            const int id  = tid + 256 * i;   // 0..511
            const int row = id >> 2;         // 0..127
            const int c   = id & 3;          // 0..3 uint4s of the k-slice
            const uint4* ga = xg + (size_t)(m0 + row) * 64 + kk * 4 + c;
            const uint4* gb = cg + (size_t)(n0 + row) * 64 + kk * 4 + c;
            uint32_t* la = &A[row * PITCH_U32 + c * 4];
            uint32_t* lb = &B[row * PITCH_U32 + c * 4];
#if HAVE_ASYNC_LDS
            __builtin_amdgcn_global_load_async_to_lds_b128((v4i*)ga, (v4i*)la, 0, 0);
            __builtin_amdgcn_global_load_async_to_lds_b128((v4i*)gb, (v4i*)lb, 0, 0);
#else
            *(uint4*)la = *ga;
            *(uint4*)lb = *gb;
#endif
        }
    };

    copy_tile(0, 0);
    wait_async_copies();
    __syncthreads();

    union FragCast { uint4 q[2]; v16bf v; };

    const int NK = D_DIM / 32;   // 16 K-iterations
    for (int kk = 0; kk < NK; ++kk) {
        const int cur = kk & 1;
        if (kk + 1 < NK) copy_tile(1 - cur, kk + 1);   // overlaps with WMMA below

        const uint32_t* A = &ldsU[cur * TILE_U32];
        const uint32_t* B = &ldsU[2 * TILE_U32 + cur * TILE_U32];

        // --- A fragments: 16x32 bf16, ISA per-lane layout -------------------
        // lane<16 : row = base+l16, K bytes [0,16) then [32,48)
        // lane>=16: row = base+l16, K bytes [16,32) then [48,64)
        v16bf afrag[4];
#pragma unroll
        for (int s = 0; s < 4; ++s) {
            const int r  = waveM * 64 + s * 16 + l16;
            const int ko = hiH * 4;                      // u32 offset (16B)
            FragCast fc;
            fc.q[0] = *(const uint4*)&A[r * PITCH_U32 + ko];
            fc.q[1] = *(const uint4*)&A[r * PITCH_U32 + ko + 8];
            afrag[s] = fc.v;
        }
        // --- B fragments: 32x16 bf16 (cb row = output col) ------------------
        // lane<16 : col = base+l16, K bytes [0,16) then [32,48) ; lane>=16: K bytes [16,32) then [48,64)
        v16bf bfrag[2];
#pragma unroll
        for (int t = 0; t < 2; ++t) {
            const int r  = waveN * 32 + t * 16 + l16;
            const int ko = hiH * 4;                      // u32 offset (16B)
            FragCast fc;
            fc.q[0] = *(const uint4*)&B[r * PITCH_U32 + ko];
            fc.q[1] = *(const uint4*)&B[r * PITCH_U32 + ko + 8];
            bfrag[t] = fc.v;
        }

#pragma unroll
        for (int s = 0; s < 4; ++s)
#pragma unroll
            for (int t = 0; t < 2; ++t)
                acc[s][t] = __builtin_amdgcn_wmma_f32_16x16x32_bf16(
                    false, afrag[s], false, bfrag[t],
                    (short)0, acc[s][t], false, false);

        if (kk + 1 < NK) wait_async_copies();  // own async stores to LDS done
        __syncthreads();                        // everyone's stores visible
    }

    // ---- epilogue: prec * (2*acc - xsq[m] - csq[n]) ------------------------
    const float prec = *prec_p;
#pragma unroll
    for (int s = 0; s < 4; ++s) {
#pragma unroll
        for (int t = 0; t < 2; ++t) {
            const int n    = n0 + waveN * 32 + t * 16 + l16;
            const float cs = csq[n];
#pragma unroll
            for (int r = 0; r < 8; ++r) {
                // C/D layout: VGPR r -> M = r (lanes 0-15) / r+8 (lanes 16-31)
                const int m   = m0 + waveM * 64 + s * 16 + r + hiH * 8;
                const float v = prec * (2.0f * acc[s][t][r] - xsq[m] - cs);
                out[(size_t)m * N_DIM + n] = v;
            }
        }
    }
}

// ---------------------------------------------------------------------------
extern "C" void kernel_launch(void* const* d_in, const int* in_sizes, int n_in,
                              void* d_out, int out_size, void* d_ws, size_t ws_size,
                              hipStream_t stream) {
    const float* x    = (const float*)d_in[0];   // [8,1024,512]
    const float* cb   = (const float*)d_in[1];   // [1024,512]
    const float* prec = (const float*)d_in[2];   // [1]
    float* out = (float*)d_out;                  // [8,1024,1024]

    // workspace layout: bf16 x (8 MB) | bf16 cb (1 MB) | xsq (32 KB) | csq (4 KB)
    uint32_t* xbf  = (uint32_t*)d_ws;                            // 8192*256 u32
    uint32_t* cbbf = xbf + (size_t)M_DIM * (D_DIM / 2);          // 1024*256 u32
    float* xsq = (float*)(cbbf + (size_t)N_DIM * (D_DIM / 2));
    float* csq = xsq + M_DIM;

    // 9216 rows / 8 rows-per-block
    prep_kernel<<<(M_DIM + N_DIM) / 8, 256, 0, stream>>>(x, cb, xbf, cbbf, xsq, csq);

    dim3 grid(N_DIM / 128, M_DIM / 128);               // (8, 64)
    vq_score_kernel<<<grid, 256, 0, stream>>>(xbf, cbbf, xsq, csq, prec, out);
}
